// TripletAggregate_71012989272669
// MI455X (gfx1250) — matrix operand order repaired and checked by reading, not compiled, not run on hardware
//
#include <hip/hip_runtime.h>
#include <hip/hip_bf16.h>

// ---------------------------------------------------------------------------
// TripletAggregate on MI455X (gfx1250, wave32, WMMA f32_16x16x32_f16,
// async global->LDS double buffering for the softmax-prob operand).
// ---------------------------------------------------------------------------
#define NB 2
#define NN 384
#define EE 64
#define HH 8
#define DD 8

typedef __attribute__((ext_vector_type(16))) _Float16 v16h;
typedef __attribute__((ext_vector_type(8)))  float    v8f;
typedef int i32x4 __attribute__((vector_size(16)));

struct alignas(16) h8 { _Float16 h[8]; };

// A-fragment (16x32 f16, M x K): lane l holds row (l&15); K halves per lane
// group: lanes 0-15 -> K in {0..15}, lanes 16-31 -> K in {16..31}.
__device__ inline v16h load_afrag(const _Float16* base, int row_stride,
                                  int row0, int k0, int lane) {
  int row = row0 + (lane & 15);
  int kh  = (lane >> 4) * 16;
  const h8* p0 = reinterpret_cast<const h8*>(base + (size_t)row * row_stride + k0 + kh);
  const h8* p1 = reinterpret_cast<const h8*>(base + (size_t)row * row_stride + k0 + 8 + kh);
  h8 a = *p0, b = *p1;
  v16h r;
#pragma unroll
  for (int e = 0; e < 8; ++e) { r[e] = a.h[e]; r[e + 8] = b.h[e]; }
  return r;
}

// B-fragment (32x16 f16, K x N): lane l holds column n0+(l&15); lanes 0-15
// carry K=0..15, lanes 16-31 carry K=16..31. Source stored [col][k].
__device__ inline v16h load_bfrag(const _Float16* base, int col_stride,
                                  int col0, int k0, int lane) {
  int col = col0 + (lane & 15);
  int kb  = (lane >> 4) * 16;
  const h8* p0 = reinterpret_cast<const h8*>(base + (size_t)col * col_stride + k0 + kb);
  const h8* p1 = p0 + 1;
  h8 a = *p0, b = *p1;
  v16h r;
#pragma unroll
  for (int e = 0; e < 8; ++e) { r[e] = a.h[e]; r[e + 8] = b.h[e]; }
  return r;
}

// 16B async copy global -> LDS (GLOBAL_LOAD_ASYNC_TO_LDS_B128, ASYNCcnt).
// Builtin takes (int4 AS1*, int4 AS3*, imm offset, imm cpol); HIP prints AS1
// as __device__ and AS3 as __shared__ in diagnostics.
__device__ inline void async_cp16(const _Float16* g, _Float16* l) {
  __builtin_amdgcn_global_load_async_to_lds_b128(
      (__attribute__((address_space(1))) i32x4*)(void*)(_Float16*)g,
      (__attribute__((address_space(3))) i32x4*)(void*)l, 0, 0);
}

// ---------------------------------------------------------------------------
// Kernel 1: fused LayerNorm + (V, EG) projections via WMMA.
//   rows r = ((b*N)+x)*N + y ; 8 waves x 16 rows = 128 rows / block.
//   Since N % 128 == 0, a block never crosses an (b,x) boundary.
//   Writes Vw[b][h][part][j][d][k] (f16) and EG[b][x][y][32] (f32).
// ---------------------------------------------------------------------------
__global__ __launch_bounds__(256) void k1_ln_proj(
    const float* __restrict__ e, const float* __restrict__ ln_s,
    const float* __restrict__ ln_b,
    const float* __restrict__ W_V, const float* __restrict__ b_V,
    const float* __restrict__ W_EG, const float* __restrict__ b_EG,
    _Float16* __restrict__ Vw, float* __restrict__ EG) {
  __shared__ _Float16 sW[160 * 64];       // weights transposed: [ch][k]
  __shared__ float    sBias[160];
  __shared__ _Float16 sE[8][16 * 64];     // per-wave e_ln rows (f16)

  const int tid = threadIdx.x;
  for (int idx = tid; idx < 160 * 64; idx += 256) {
    int ch = idx >> 6, kk = idx & 63;
    float w = (ch < 128) ? W_V[kk * 128 + ch] : W_EG[kk * 32 + (ch - 128)];
    sW[idx] = (_Float16)w;
  }
  if (tid < 160) sBias[tid] = (tid < 128) ? b_V[tid] : b_EG[tid - 128];

  const int wave = tid >> 5, lane = tid & 31;

  // Scalar decode of (b, x, y0): block-uniform (SALU), no per-lane 64-bit math.
  const int base = blockIdx.x * 128;       // flat row of block start
  const int bx   = base / NN;              // b*N + x
  const int bb   = bx / NN;
  const int x    = bx - bb * NN;
  const int y0   = base - bx * NN;         // multiple of 128
  const size_t row0 = (size_t)base + (size_t)wave * 16;

  // LayerNorm: lane pair (l, l+16) handles one row, 32 channels each.
  {
    int r = lane & 15;
    int ch0 = (lane >> 4) * 32;
    const float* ep = e + (row0 + r) * 64 + ch0;
    float v[32];
    float s = 0.f, s2 = 0.f;
#pragma unroll
    for (int c = 0; c < 32; ++c) { v[c] = ep[c]; s += v[c]; s2 += v[c] * v[c]; }
    s  += __shfl_xor(s, 16);
    s2 += __shfl_xor(s2, 16);
    float mu   = s * (1.f / 64.f);
    float var  = s2 * (1.f / 64.f) - mu * mu;
    float rstd = rsqrtf(var + 1e-5f);
#pragma unroll
    for (int c = 0; c < 32; ++c) {
      float g = ln_s[ch0 + c], bb2 = ln_b[ch0 + c];
      sE[wave][r * 64 + ch0 + c] = (_Float16)((v[c] - mu) * rstd * g + bb2);
    }
  }
  __syncthreads();

  v8f acc[10];
#pragma unroll
  for (int n = 0; n < 10; ++n) { v8f z = {}; acc[n] = z; }

#pragma unroll
  for (int s = 0; s < 2; ++s) {
    v16h a = load_afrag(&sE[wave][0], 64, 0, s * 32, lane);
#pragma unroll
    for (int n = 0; n < 10; ++n) {
      v16h bf = load_bfrag(sW, 64, n * 16, s * 32, lane);
      acc[n] = __builtin_amdgcn_wmma_f32_16x16x32_f16(
          false, a, false, bf, (short)0, acc[n], false, false);
    }
  }

#pragma unroll
  for (int n = 0; n < 10; ++n) {
    int ch = n * 16 + (lane & 15);
    float bias = sBias[ch];
#pragma unroll
    for (int v = 0; v < 8; ++v) {
      int m = (lane >> 4) * 8 + v;
      int y = y0 + wave * 16 + m;
      float val = acc[n][v] + bias;
      if (ch < 128) {
        int part = ch >> 6, dd = (ch >> 3) & 7, hh = ch & 7;
        size_t bas = (((size_t)bb * HH + hh) * 2 + part) * ((size_t)NN * DD * NN);
        size_t off = (part == 0)
                         ? bas + ((size_t)x * DD + dd) * NN + y   // j=x, k=y
                         : bas + ((size_t)y * DD + dd) * NN + x;  // j=y, k=x
        Vw[off] = (_Float16)val;
      } else {
        EG[(((size_t)bb * NN + x) * NN + y) * 32 + (ch - 128)] = val;
      }
    }
  }
}

// ---------------------------------------------------------------------------
// Kernel 2: gated softmax over k. One wave per (b,i,h); grid.y = dir (0:in,1:out).
//   Writes P[dir][b][h][i][k] f16 = softmax_k(E) * sigmoid(G).
// ---------------------------------------------------------------------------
__global__ __launch_bounds__(256) void k2_softmax(
    const float* __restrict__ EG, const float* __restrict__ mask,
    _Float16* __restrict__ P) {
  const int wave = threadIdx.x >> 5, lane = threadIdx.x & 31;
  const int task = blockIdx.x * 8 + wave;  // over B*N*H
  const int dir  = blockIdx.y;
  const int h  = task & 7;
  const int bi = task >> 3;
  const int i  = bi % NN;
  const int b  = bi / NN;

  float lg[12], gt[12];
  float mx = -3.4e38f;
#pragma unroll
  for (int t = 0; t < 12; ++t) {
    int k = lane + 32 * t;
    if (dir == 0) {  // incoming: logits E_in[b,i,k,h] + mask[b,i,k]
      size_t cell = (((size_t)b * NN + i) * NN + k) * 32;
      float mval  = mask[((size_t)b * NN + i) * NN + k];
      lg[t] = EG[cell + h] + mval;
      gt[t] = EG[cell + 8 + h] + mval;
    } else {         // outgoing: logits E_out[b,k,i,h] (no mask, per reference)
      size_t cell = (((size_t)b * NN + k) * NN + i) * 32;
      lg[t] = EG[cell + 16 + h];
      gt[t] = EG[cell + 24 + h];
    }
    mx = fmaxf(mx, lg[t]);
  }
#pragma unroll
  for (int o = 16; o >= 1; o >>= 1) mx = fmaxf(mx, __shfl_xor(mx, o));
  float sum = 0.f;
#pragma unroll
  for (int t = 0; t < 12; ++t) { lg[t] = __expf(lg[t] - mx); sum += lg[t]; }
#pragma unroll
  for (int o = 16; o >= 1; o >>= 1) sum += __shfl_xor(sum, o);
  float inv = 1.f / sum;

  const size_t pbase = ((((size_t)dir * NB + b) * HH + h) * NN + i) * (size_t)NN;
#pragma unroll
  for (int t = 0; t < 12; ++t) {
    int k = lane + 32 * t;
    float sg = 1.f / (1.f + __expf(-gt[t]));
    P[pbase + k] = (_Float16)(lg[t] * inv * sg);
  }
}

// ---------------------------------------------------------------------------
// Kernel 3: both einsums + fused W_O projection.
//   One workgroup per (b, i16, j16) tile; wave w = head h.
//   Phase 1: C[jd,i] = sum_k V[jd,k] * P[k,i] for in & out parts (WMMA, K=32),
//            with the P tiles async-copied global->LDS, double buffered on
//            ASYNCcnt, and the V stream prefetched.
//   Phase 2: stage Va tile (f16) in LDS (disjoint channels per wave), barrier,
//            project 256x128 @ 128x64 with WMMA, add b_O, store fp32.
// ---------------------------------------------------------------------------

// Issue async copy of P tile: rows i0..i0+15, halfs k..k+31 -> lbase[row*32+kk].
// 64 chunks of 16B, 32 lanes -> 2 async instructions.
__device__ inline void issue_ptile(const _Float16* gbase, _Float16* lbase,
                                   int i0, int k, int lane) {
#pragma unroll
  for (int q = 0; q < 2; ++q) {
    int c   = q * 32 + lane;   // chunk id 0..63
    int row = c >> 2;
    int seg = c & 3;           // 16B segment within the 64B row
    async_cp16(gbase + (size_t)(i0 + row) * NN + k + seg * 8,
               lbase + row * 32 + seg * 8);
  }
}

__global__ __launch_bounds__(256) void k3_agg_proj(
    const _Float16* __restrict__ Vw, const _Float16* __restrict__ P,
    const float* __restrict__ W_O, const float* __restrict__ b_O,
    float* __restrict__ out) {
  __shared__ _Float16 sVa[16 * 16 * 128];     // [i_local][j_local][ch] = 64 KB
  __shared__ _Float16 sP[8][2][2][16 * 32];   // [wave][buf][part][.] = 32 KB

  const int wave = threadIdx.x >> 5, lane = threadIdx.x & 31;
  const int h  = wave;
  const int i0 = blockIdx.x * 16, j0 = blockIdx.y * 16, b = blockIdx.z;

  const size_t PS = (size_t)NN * NN;
  const _Float16* Pin  = P + (((size_t)0 * NB + b) * HH + h) * PS;
  const _Float16* Pout = P + (((size_t)1 * NB + b) * HH + h) * PS;
  const size_t VS = (size_t)NN * DD * NN;
  const _Float16* Vin  = Vw + (((size_t)b * HH + h) * 2 + 0) * VS;
  const _Float16* Vout = Vw + (((size_t)b * HH + h) * 2 + 1) * VS;

  v8f accI[8], accO[8];
#pragma unroll
  for (int t = 0; t < 8; ++t) { v8f z = {}; accI[t] = z; accO[t] = z; }

  // Prologue: async-stage P tiles for k = 0 into buffer 0.
  issue_ptile(Pin,  &sP[wave][0][0][0], i0, 0, lane);
  issue_ptile(Pout, &sP[wave][0][1][0], i0, 0, lane);

  for (int s = 0; s < 12; ++s) {
    const int k   = s * 32;
    const int buf = s & 1;
    if (s + 1 < 12) {
      // Stage next k-slab into the other buffer, then wait until the 4 async
      // ops of the *current* buffer have completed (in-order completion).
      issue_ptile(Pin,  &sP[wave][buf ^ 1][0][0], i0, k + 32, lane);
      issue_ptile(Pout, &sP[wave][buf ^ 1][1][0], i0, k + 32, lane);
      __builtin_amdgcn_s_wait_asynccnt(4);
    } else {
      __builtin_amdgcn_s_wait_asynccnt(0);
    }

    // Prefetch next k-slab of the V stream (global_prefetch).
    if (s + 1 < 12) {
      __builtin_prefetch(Vin  + ((size_t)j0 * DD + (lane & 15)) * NN + k + 32, 0, 1);
      __builtin_prefetch(Vout + ((size_t)j0 * DD + (lane & 15)) * NN + k + 32, 0, 1);
    }

    v16h bIn  = load_bfrag(&sP[wave][buf][0][0], 32, 0, 0, lane);
    v16h bOut = load_bfrag(&sP[wave][buf][1][0], 32, 0, 0, lane);
#pragma unroll
    for (int t = 0; t < 8; ++t) {
      v16h aI = load_afrag(Vin, NN, j0 * DD + t * 16, k, lane);
      accI[t] = __builtin_amdgcn_wmma_f32_16x16x32_f16(
          false, aI, false, bIn, (short)0, accI[t], false, false);
      v16h aO = load_afrag(Vout, NN, j0 * DD + t * 16, k, lane);
      accO[t] = __builtin_amdgcn_wmma_f32_16x16x32_f16(
          false, aO, false, bOut, (short)0, accO[t], false, false);
    }
  }

  // Stage Va tile: wave w owns channels {part*64 + d*8 + w} -> disjoint writes.
#pragma unroll
  for (int t = 0; t < 8; ++t) {
#pragma unroll
    for (int v = 0; v < 8; ++v) {
      int m  = (lane >> 4) * 8 + v;      // row within jd tile
      int il = lane & 15;                // C/D: lane = N = i_local
      int jd = t * 16 + m;
      int jl = jd >> 3, dd = jd & 7;
      sVa[(il * 16 + jl) * 128 + dd * 8 + h]      = (_Float16)accI[t][v];
      sVa[(il * 16 + jl) * 128 + 64 + dd * 8 + h] = (_Float16)accO[t][v];
    }
  }
  __syncthreads();

  // Projection: rows = (i_local,j_local) [256], K = 128 channels, N = 64 out.
  v8f o[2][4];
#pragma unroll
  for (int mt = 0; mt < 2; ++mt)
#pragma unroll
    for (int n = 0; n < 4; ++n) { v8f z = {}; o[mt][n] = z; }

#pragma unroll
  for (int kk = 0; kk < 4; ++kk) {
    v16h bf[4];
#pragma unroll
    for (int n = 0; n < 4; ++n) {
      int c  = n * 16 + (lane & 15);
      int kb = (lane >> 4) * 16;
      v16h r;
#pragma unroll
      for (int e2 = 0; e2 < 16; ++e2)
        r[e2] = (_Float16)W_O[(size_t)(kk * 32 + kb + e2) * EE + c];
      bf[n] = r;
    }
#pragma unroll
    for (int mt = 0; mt < 2; ++mt) {
      v16h a = load_afrag(sVa, 128, wave * 32 + mt * 16, kk * 32, lane);
#pragma unroll
      for (int n = 0; n < 4; ++n)
        o[mt][n] = __builtin_amdgcn_wmma_f32_16x16x32_f16(
            false, a, false, bf[n], (short)0, o[mt][n], false, false);
    }
  }

#pragma unroll
  for (int mt = 0; mt < 2; ++mt)
#pragma unroll
    for (int n = 0; n < 4; ++n) {
      int c = n * 16 + (lane & 15);
      float bo = b_O[c];
#pragma unroll
      for (int v = 0; v < 8; ++v) {
        int m  = (lane >> 4) * 8 + v;
        int rr = wave * 32 + mt * 16 + m;
        int il = rr >> 4, jl = rr & 15;
        out[(((size_t)b * NN + i0 + il) * NN + (j0 + jl)) * EE + c] =
            o[mt][n][v] + bo;
      }
    }
}

// ---------------------------------------------------------------------------
extern "C" void kernel_launch(void* const* d_in, const int* in_sizes, int n_in,
                              void* d_out, int out_size, void* d_ws, size_t ws_size,
                              hipStream_t stream) {
  (void)in_sizes; (void)n_in; (void)out_size; (void)ws_size;
  const float* e    = (const float*)d_in[0];
  const float* mask = (const float*)d_in[1];
  const float* ln_s = (const float*)d_in[2];
  const float* ln_b = (const float*)d_in[3];
  const float* W_V  = (const float*)d_in[4];
  const float* b_V  = (const float*)d_in[5];
  const float* W_EG = (const float*)d_in[6];
  const float* b_EG = (const float*)d_in[7];
  const float* W_O  = (const float*)d_in[8];
  const float* b_O  = (const float*)d_in[9];
  float* out = (float*)d_out;

  // Workspace layout:
  //   Vw : f16 [B][H][2][N][D][N]  = 75,497,472 B
  //   EG : f32 [B][N][N][32]       = 37,748,736 B
  //   P  : f16 [2][B][H][N][N]     =  9,437,184 B   (total ~122.7 MB)
  char* ws = (char*)d_ws;
  _Float16* Vw = (_Float16*)ws;
  float*    EG = (float*)(ws + 75497472ull);
  _Float16* P  = (_Float16*)(ws + 75497472ull + 37748736ull);

  const size_t rows = (size_t)NB * NN * NN;             // 294912
  k1_ln_proj<<<dim3((unsigned)(rows / 128)), 256, 0, stream>>>(
      e, ln_s, ln_b, W_V, b_V, W_EG, b_EG, Vw, EG);

  k2_softmax<<<dim3(NB * NN * HH / 8, 2), 256, 0, stream>>>(EG, mask, P);

  k3_agg_proj<<<dim3(NN / 16, NN / 16, NB), 256, 0, stream>>>(
      Vw, P, W_O, b_O, out);
}
